// LSTMFrame_18872086298648
// MI455X (gfx1250) — matrix operand order repaired and checked by reading, not compiled, not run on hardware
//
#include <hip/hip_runtime.h>
#include <hip/hip_bf16.h>
#include <stdint.h>

// ---------------------------------------------------------------------------
// Types for CDNA5 WMMA / TDM (gfx1250, wave32)
// ---------------------------------------------------------------------------
typedef __attribute__((ext_vector_type(16))) __bf16 bf16x16;
typedef __attribute__((ext_vector_type(8)))  float  f32x8;
typedef __attribute__((ext_vector_type(4)))  unsigned int uint32x4;
typedef __attribute__((ext_vector_type(8)))  int    int32x8;
typedef __attribute__((ext_vector_type(4)))  int    int32x4;

struct Bits32B { uint4 lo; uint4 hi; };

__device__ __forceinline__ unsigned short f2bf(float f) {
    unsigned int u = __float_as_uint(f);
    unsigned int r = (u + 0x7FFFu + ((u >> 16) & 1u)) >> 16;
    return (unsigned short)r;
}

__device__ __forceinline__ float sigmoidf_fast(float x) {
    return 1.0f / (1.0f + __expf(-x));
}

// A-fragment (16x32 bf16): per-lane two contiguous 8-elem halves at +0 and +8.
__device__ __forceinline__ bf16x16 load_frag_2x16(const unsigned short* p0,
                                                  const unsigned short* p1) {
    Bits32B t;
    t.lo = *(const uint4*)p0;
    t.hi = *(const uint4*)p1;
    return __builtin_bit_cast(bf16x16, t);
}

// B-fragment (32x16 bf16): per-lane one contiguous 16-elem (32B) chunk.
__device__ __forceinline__ bf16x16 load_frag_32b(const unsigned short* p) {
    Bits32B t = *(const Bits32B*)p;
    return __builtin_bit_cast(bf16x16, t);
}

__device__ __forceinline__ f32x8 wmma_bf16(bf16x16 a, bf16x16 b, f32x8 c) {
    return __builtin_amdgcn_wmma_f32_16x16x32_bf16(
        false, a, false, b, (short)0, c, false, false);
}

// ---------------------------------------------------------------------------
// Tensor Data Mover: 2D tile (rows x cols, bf16) global -> LDS.
// D# layout per CDNA5 ISA.
// Issue from ONE wave; caller barriers afterwards.
// ---------------------------------------------------------------------------
#if __has_builtin(__builtin_amdgcn_tensor_load_to_lds)
#define HAVE_TDM 1
#else
#define HAVE_TDM 0
#endif

__device__ __forceinline__ void tdm_load_tile_bf16(unsigned lds_byte_addr,
                                                   const void* gptr,
                                                   unsigned rows,
                                                   unsigned cols,
                                                   unsigned stride_elems) {
#if HAVE_TDM
    const uint64_t ga = (uint64_t)(uintptr_t)gptr;
    uint32x4 g0;
    g0[0] = 1u;                                   // count=1, user descriptor
    g0[1] = lds_byte_addr;                        // lds_addr
    g0[2] = (unsigned)(ga & 0xFFFFFFFFu);         // global_addr[31:0]
    g0[3] = (unsigned)((ga >> 32) & 0x01FFFFFFu)  // global_addr[56:32]
            | (2u << 30);                         // type=2 ("image")

    int32x8 g1;
    g1[0] = (int)(1u << 16);                      // data_size=1 -> 2 bytes
    g1[1] = (int)((cols & 0xFFFFu) << 16);        // tensor_dim0[15:0]
    g1[2] = (int)(((cols >> 16) & 0xFFFFu) | ((rows & 0xFFFFu) << 16)); // td0 hi | td1 lo
    g1[3] = (int)(((rows >> 16) & 0xFFFFu) | ((cols & 0xFFFFu) << 16)); // td1 hi | tile_dim0
    g1[4] = (int)(rows & 0xFFFFu);                // tile_dim1 (tile_dim2=0)
    g1[5] = (int)stride_elems;                    // tensor_dim0_stride[31:0]
    g1[6] = 0;                                    // stride hi | dim1_stride lo
    g1[7] = 0;

    int32x4 gz = {0, 0, 0, 0};
#if defined(__clang_major__) && (__clang_major__ >= 23)
    int32x8 gz8 = {0, 0, 0, 0, 0, 0, 0, 0};
    __builtin_amdgcn_tensor_load_to_lds(g0, g1, gz, gz, gz8, 0);
#else
    __builtin_amdgcn_tensor_load_to_lds(g0, g1, gz, gz, 0);
#endif
    __builtin_amdgcn_s_wait_tensorcnt(0);
#else
    (void)lds_byte_addr; (void)gptr; (void)rows; (void)cols; (void)stride_elems;
#endif
}

// ---------------------------------------------------------------------------
// Problem constants
// ---------------------------------------------------------------------------
constexpr int T_   = 2048;
constexpr int B_   = 64;
constexpr int D_   = 256;
constexpr int H_   = 256;
constexpr int G4H  = 1024;        // 4*H
constexpr int TB   = T_ * B_;     // 131072 rows

// ---------------------------------------------------------------------------
// fp32 -> bf16 pack / bias sum
// ---------------------------------------------------------------------------
__global__ void pack_bf16_kernel(const float* __restrict__ src,
                                 unsigned short* __restrict__ dst, size_t n) {
    size_t i = (size_t)blockIdx.x * blockDim.x + threadIdx.x;
    size_t stride = (size_t)gridDim.x * blockDim.x;
    for (; i < n; i += stride) dst[i] = f2bf(src[i]);
}

__global__ void bias_sum_kernel(const float* __restrict__ a,
                                const float* __restrict__ b,
                                float* __restrict__ o, int n) {
    int i = blockIdx.x * blockDim.x + threadIdx.x;
    if (i < n) o[i] = a[i] + b[i];
}

// ---------------------------------------------------------------------------
// Input projection GEMM:  Z[m, n] = sum_k A[m, k] * W[n, k] + bias[n]
// Block = 16 waves covering a 64(M) x 1024(N) strip; the shared 64 x K A-tile
// is staged in LDS via the Tensor Data Mover (issued by wave 0), then each
// wave computes 4x4 = 16 accumulator tiles (~1 b128 load per WMMA).
// ---------------------------------------------------------------------------
__global__ void __launch_bounds__(512, 1)
wmma_inproj_kernel(const unsigned short* __restrict__ A, int K,
                   const unsigned short* __restrict__ W,
                   const float* __restrict__ bias,
                   float* __restrict__ Z) {
    __shared__ unsigned short a_sh[64 * 512];     // up to K=512 (64 KB)

    const int lane = threadIdx.x & 31;
    const int wave = threadIdx.x >> 5;
    const int lhi = lane >> 4, llo = lane & 15;
    const int mBase = blockIdx.x * 64;
    const int nBase = wave * 64;

    // ---- stage A tile (64 rows x K) into LDS ----
#if HAVE_TDM
    if (threadIdx.x < 32) {
        tdm_load_tile_bf16((unsigned)(uintptr_t)&a_sh[0],
                           A + (size_t)mBase * K, 64u, (unsigned)K, (unsigned)K);
    }
#else
    {   // cooperative copy fallback: tile is a contiguous 64*K*2-byte block
        const uint4* src = (const uint4*)(A + (size_t)mBase * K);
        uint4* dst = (uint4*)a_sh;
        const int nchunks = (64 * K) / 8;
        for (int i = threadIdx.x; i < nchunks; i += 512) dst[i] = src[i];
    }
#endif
    __syncthreads();

    f32x8 acc[4][4];
#pragma unroll
    for (int mt = 0; mt < 4; ++mt)
#pragma unroll
        for (int nt = 0; nt < 4; ++nt)
            acc[mt][nt] = f32x8{0.f,0.f,0.f,0.f,0.f,0.f,0.f,0.f};

    const int nkt = K >> 5;
    for (int kt = 0; kt < nkt; ++kt) {
        bf16x16 bfrag[4];
#pragma unroll
        for (int nt = 0; nt < 4; ++nt) {
            const unsigned short* Wp =
                W + (size_t)(nBase + nt * 16 + llo) * K + kt * 32 + lhi * 16;
            bfrag[nt] = load_frag_32b(Wp);
        }
#pragma unroll
        for (int mt = 0; mt < 4; ++mt) {
            const unsigned short* ap =
                &a_sh[(mt * 16 + llo) * K + kt * 32 + lhi * 16];
            bf16x16 a = load_frag_2x16(ap, ap + 8);
#pragma unroll
            for (int nt = 0; nt < 4; ++nt)
                acc[mt][nt] = wmma_bf16(a, bfrag[nt], acc[mt][nt]);
        }
    }
#pragma unroll
    for (int nt = 0; nt < 4; ++nt) {
        const int n = nBase + nt * 16 + llo;
        const float bv = bias[n];
#pragma unroll
        for (int mt = 0; mt < 4; ++mt) {
#pragma unroll
            for (int r = 0; r < 8; ++r) {
                const int m = mBase + mt * 16 + lhi * 8 + r;
                Z[(size_t)m * G4H + n] = acc[mt][nt][r] + bv;
            }
        }
    }
}

// ---------------------------------------------------------------------------
// Persistent recurrent kernel. LSTM state is batch-parallel, so each cell is
// split over 4 workgroups of 16 batch rows (no cross-WG traffic at all),
// quartering the sequential WMMA chain per step.
//   grid.x = 8: dir = blockIdx.x & 1, batch group = blockIdx.x >> 1.
// Each of the 16 waves owns one 16-wide h-slice for ALL four gates, so the
// gate nonlinearity and the c-state stay entirely in registers; h lives in
// 8 KB of LDS; W_hh streams from L2 directly in WMMA B-fragment layout.
// ---------------------------------------------------------------------------
__global__ void __launch_bounds__(512, 1)
lstm_recurrent_kernel(const float* __restrict__ ZxF,
                      const float* __restrict__ ZxB,
                      const unsigned short* __restrict__ WhhF,
                      const unsigned short* __restrict__ WhhB,
                      const int* __restrict__ lengths,
                      unsigned short* __restrict__ out_bf,  // layer0 out (T,B,512) bf16
                      float* __restrict__ out_f,            // layer1 out (T,B,512) f32
                      float* __restrict__ last_h,
                      float* __restrict__ last_c,
                      int cellBase) {
    const int dir = blockIdx.x & 1;
    const int mBase = (blockIdx.x >> 1) * 16;      // this WG's 16 batch rows
    const float* Zx = dir ? ZxB : ZxF;
    const unsigned short* Whh = dir ? WhhB : WhhF;

    const int lane = threadIdx.x & 31;
    const int wave = threadIdx.x >> 5;
    const int lhi = lane >> 4, llo = lane & 15;
    const int hbase = wave * 16;
    const int hcol  = hbase + llo;
    const int cell  = cellBase + dir;

    __shared__ unsigned short h_sh[16 * H_];       // 16 x 256 bf16 (8 KB)
    __shared__ int len_s[16];

    for (int i = threadIdx.x; i < 16 * H_; i += 512) h_sh[i] = 0;
    if (threadIdx.x < 16) len_s[threadIdx.x] = lengths[mBase + threadIdx.x];
    __syncthreads();

    f32x8 cst = f32x8{0.f,0.f,0.f,0.f,0.f,0.f,0.f,0.f};

    for (int s = 0; s < T_; ++s) {
        // ---- seed accumulators with precomputed input projection ----
        f32x8 acc[4];
#pragma unroll
        for (int g = 0; g < 4; ++g) {
            const int n = g * H_ + hcol;
#pragma unroll
            for (int r = 0; r < 8; ++r) {
                const int ml = lhi * 8 + r;
                int t = dir ? (len_s[ml] - 1 - s) : s;
                if (t < 0) t += T_;
                acc[g][r] = Zx[((size_t)t * B_ + mBase + ml) * G4H + n];
            }
        }
        // prefetch next step's Zx rows (global_prefetch_b8)
        if (s + 1 < T_) {
            const int ml0 = lhi * 8;
            int tn = dir ? (len_s[ml0] - 2 - s) : (s + 1);
            if (tn < 0) tn += T_;
#pragma unroll
            for (int g = 0; g < 4; ++g)
                __builtin_prefetch(&Zx[((size_t)tn * B_ + mBase + ml0) * G4H +
                                       g * H_ + hcol], 0, 3);
        }
        // ---- z += h @ Whh^T via WMMA ----
        for (int kt = 0; kt < 8; ++kt) {
            const unsigned short* ap = &h_sh[llo * H_ + kt * 32 + lhi * 16];
            bf16x16 a = load_frag_2x16(ap, ap + 8);
#pragma unroll
            for (int g = 0; g < 4; ++g) {
                const unsigned short* Wp =
                    Whh + (size_t)(g * H_ + hbase + llo) * H_ + kt * 32 + lhi * 16;
                acc[g] = wmma_bf16(a, load_frag_32b(Wp), acc[g]);
            }
        }
        __syncthreads();   // all h_sh reads for this step complete

        // ---- gates / state update (register-local per wave) ----
#pragma unroll
        for (int r = 0; r < 8; ++r) {
            const int ml = lhi * 8 + r;
            const int m  = mBase + ml;
            const int len = len_s[ml];
            int t = dir ? (len - 1 - s) : s;
            if (t < 0) t += T_;
            const float zi = acc[0][r], zf = acc[1][r];
            const float zg = acc[2][r], zo = acc[3][r];
            const float c = sigmoidf_fast(zf) * cst[r] +
                            sigmoidf_fast(zi) * tanhf(zg);
            const float h = sigmoidf_fast(zo) * tanhf(c);
            cst[r] = c;

            h_sh[ml * H_ + hcol] = f2bf(h);

            const size_t orow = ((size_t)t * B_ + m) * 512 + dir * H_ + hcol;
            if (out_bf) out_bf[orow] = f2bf(h);
            else        out_f[orow] = (t < len) ? h : 0.0f;

            if (s == len - 1) {
                last_h[((size_t)cell * B_ + m) * H_ + hcol] = h;
                last_c[((size_t)cell * B_ + m) * H_ + hcol] = c;
            }
        }
        __syncthreads();   // new h visible before next step's A-fragment reads
    }
}

// ---------------------------------------------------------------------------
// Host-side orchestration
// ---------------------------------------------------------------------------
extern "C" void kernel_launch(void* const* d_in, const int* in_sizes, int n_in,
                              void* d_out, int out_size, void* d_ws, size_t ws_size,
                              hipStream_t stream) {
    (void)in_sizes; (void)n_in; (void)out_size; (void)ws_size;

    const float* x       = (const float*)d_in[0];
    const int*   lengths = (const int*)d_in[1];
    // cells in order [l0 fwd, l0 bwd, l1 fwd, l1 bwd]
    const float* w_ih[4] = {(const float*)d_in[2],  (const float*)d_in[6],
                            (const float*)d_in[10], (const float*)d_in[14]};
    const float* w_hh[4] = {(const float*)d_in[3],  (const float*)d_in[7],
                            (const float*)d_in[11], (const float*)d_in[15]};
    const float* b_ih[4] = {(const float*)d_in[4],  (const float*)d_in[8],
                            (const float*)d_in[12], (const float*)d_in[16]};
    const float* b_hh[4] = {(const float*)d_in[5],  (const float*)d_in[9],
                            (const float*)d_in[13], (const float*)d_in[17]};
    const size_t din[4] = {(size_t)D_, (size_t)D_, (size_t)(2 * H_), (size_t)(2 * H_)};

    // ---- workspace carve-out ----
    char* p = (char*)d_ws;
    auto alloc = [&](size_t bytes) -> char* {
        char* r = p;
        p += (bytes + 255) & ~(size_t)255;
        return r;
    };
    unsigned short* x_bf    = (unsigned short*)alloc((size_t)TB * D_ * 2);
    unsigned short* out0_bf = (unsigned short*)alloc((size_t)TB * 2 * H_ * 2);
    unsigned short* wih_bf[4];
    unsigned short* whh_bf[4];
    float* bias[4];
    for (int c = 0; c < 4; ++c) {
        wih_bf[c] = (unsigned short*)alloc((size_t)G4H * din[c] * 2);
        whh_bf[c] = (unsigned short*)alloc((size_t)G4H * H_ * 2);
        bias[c]   = (float*)alloc((size_t)G4H * 4);
    }
    float* ZxF = (float*)alloc((size_t)TB * G4H * 4);
    float* ZxB = (float*)alloc((size_t)TB * G4H * 4);

    float* out_f32 = (float*)d_out;                      // (T, B, 512)
    float* last_h  = out_f32 + (size_t)TB * 2 * H_;      // (4, B, H)
    float* last_c  = last_h + (size_t)4 * B_ * H_;       // (4, B, H)

    // ---- pack fp32 -> bf16 ----
    pack_bf16_kernel<<<4096, 256, 0, stream>>>(x, x_bf, (size_t)TB * D_);
    for (int c = 0; c < 4; ++c) {
        pack_bf16_kernel<<<512, 256, 0, stream>>>(w_ih[c], wih_bf[c],
                                                  (size_t)G4H * din[c]);
        pack_bf16_kernel<<<512, 256, 0, stream>>>(w_hh[c], whh_bf[c],
                                                  (size_t)G4H * H_);
        bias_sum_kernel<<<4, 256, 0, stream>>>(b_ih[c], b_hh[c], bias[c], G4H);
    }

    const int gemm_blocks = TB / 64;   // 2048 blocks; each covers 64 rows x full N

    // ---- layer 0 ----
    wmma_inproj_kernel<<<gemm_blocks, 512, 0, stream>>>(x_bf, D_, wih_bf[0], bias[0], ZxF);
    wmma_inproj_kernel<<<gemm_blocks, 512, 0, stream>>>(x_bf, D_, wih_bf[1], bias[1], ZxB);
    lstm_recurrent_kernel<<<8, 512, 0, stream>>>(
        ZxF, ZxB, whh_bf[0], whh_bf[1], lengths,
        out0_bf, nullptr, last_h, last_c, /*cellBase=*/0);

    // ---- layer 1 (input = layer-0 concat [fwd, bwd], Din = 512) ----
    wmma_inproj_kernel<<<gemm_blocks, 512, 0, stream>>>(out0_bf, 2 * H_, wih_bf[2], bias[2], ZxF);
    wmma_inproj_kernel<<<gemm_blocks, 512, 0, stream>>>(out0_bf, 2 * H_, wih_bf[3], bias[3], ZxB);
    lstm_recurrent_kernel<<<8, 512, 0, stream>>>(
        ZxF, ZxB, whh_bf[2], whh_bf[3], lengths,
        nullptr, out_f32, last_h, last_c, /*cellBase=*/2);
}
